// GraphAttentionNetwork_45664092291725
// MI455X (gfx1250) — matrix operand order repaired and checked by reading, not compiled, not run on hardware
//
#include <hip/hip_runtime.h>
#include <hip/hip_bf16.h>
#include <math.h>

#define N_NODES 50000
#define N_EDGES 800000
#define ETOT (N_EDGES + N_NODES)
#define F_IN 64
#define HID 128
#define HEADS 4
#define OUT_DIM 256
#define NEG_SLOPE 0.2f

typedef _Float16 half_t;
typedef __attribute__((ext_vector_type(16))) _Float16 v16h;
typedef __attribute__((ext_vector_type(8)))  float    v8f;

union Frag16 { v16h v; unsigned int u[8]; };
union Half4  { half_t h[4]; unsigned long long q; };
union Half2  { half_t h[2]; unsigned int u; };

// ---------------------------------------------------------------------------
// WMMA GEMM: C[M x Nt] = A[M x K] * B[K x Nt], fp32 in/out, f16 WMMA compute.
// Block tile 128x64, BK=32, 8 waves (4 M-slabs x 2 N-slabs), each wave 32x32.
// Branch-free clamped loads (OOB rows feed acc tiles that are never stored).
// ---------------------------------------------------------------------------
#define BM 128
#define BN 64
#define BK 32
#define AS_STR 36   // halfs; 72B row stride -> 8B-aligned b64 stores, even for pair reads
#define BS_STR 34   // halfs; 68B col stride -> 4B-aligned b32 stores/pair reads

__global__ __launch_bounds__(256) void gat_wmma_gemm(
    const float* __restrict__ A, const float* __restrict__ B,
    float* __restrict__ C, int M, int K, int Nt)
{
  __shared__ __align__(16) half_t As[BM * AS_STR];
  __shared__ __align__(16) half_t Bs[BN * BS_STR];

  const int tid  = threadIdx.x;
  const int lane = tid & 31;
  const int wave = tid >> 5;       // 0..7
  const int wm   = wave >> 1;      // 0..3  (32-row slab)
  const int wn   = wave & 1;       // 0..1  (32-col slab)
  const int grp  = lane >> 4;      // 0/1 half-wave group
  const int l16  = lane & 15;

  const int row0 = blockIdx.y * BM;
  const int col0 = blockIdx.x * BN;

  v8f acc[2][2] = {};

  const int ksteps = K / BK;
  for (int kt = 0; kt < ksteps; ++kt) {
    const int kbase = kt * BK;

    // speculative prefetch of next A k-slice (branch-free; clamped at end)
    {
      int r  = tid >> 3;
      int gr = row0 + r; gr = gr < M ? gr : M - 1;
      int kn = (kbase + BK < K) ? (kbase + BK) : kbase;
      __builtin_prefetch(&A[(size_t)gr * K + kn], 0, 3);
    }

    // A tile: 128x32 f32 -> LDS f16 row-major. 4x float4 per thread,
    // clamped row (no divergent guards), b64 LDS stores.
    #pragma unroll
    for (int i = 0; i < (BM * BK) / (256 * 4); ++i) {   // 4 iters
      int lin = tid + i * 256;                          // float4 index
      int r  = lin >> 3;                                // 8 float4 per row
      int c4 = (lin & 7) * 4;
      int gr = row0 + r; gr = gr < M ? gr : M - 1;      // clamp: branch-free
      const float4 v = *(const float4*)&A[(size_t)gr * K + kbase + c4];
      Half4 pk;
      pk.h[0] = (half_t)v.x; pk.h[1] = (half_t)v.y;
      pk.h[2] = (half_t)v.z; pk.h[3] = (half_t)v.w;
      *(unsigned long long*)&As[r * AS_STR + c4] = pk.q;
    }
    // B tile: 32x64 f32 -> LDS f16 transposed (Bs[col][k]); each thread packs
    // one K-pair (both rows coalesced across threads) into one b32 store.
    #pragma unroll
    for (int i = 0; i < (BK * BN) / (256 * 2); ++i) {   // 4 iters
      int lin = tid + i * 256;                          // pair index
      int col = lin & 63;
      int kp  = lin >> 6;                               // 0..15
      float lo = B[(size_t)(kbase + 2 * kp)     * Nt + col0 + col];
      float hi = B[(size_t)(kbase + 2 * kp + 1) * Nt + col0 + col];
      Half2 pk; pk.h[0] = (half_t)lo; pk.h[1] = (half_t)hi;
      *(unsigned int*)&Bs[col * BS_STR + 2 * kp] = pk.u;
    }
    __syncthreads();

    // Fragment loads per ISA 7.12.2 wave32 16-bit layouts (packed-pair reads).
    Frag16 fa[2], fb[2];
    #pragma unroll
    for (int i = 0; i < 2; ++i) {
      int arow = wm * 32 + i * 16 + l16;
      #pragma unroll
      for (int p = 0; p < 8; ++p) {
        // A 16x32: lanes 0-15 -> K {0..7,16..23}; lanes 16-31 -> +8
        int kk = ((p < 4) ? (2 * p) : (2 * p + 8)) + grp * 8;
        fa[i].u[p] = *(const unsigned int*)&As[arow * AS_STR + kk];
      }
    }
    #pragma unroll
    for (int j = 0; j < 2; ++j) {
      int bcol = wn * 32 + j * 16 + l16;
      #pragma unroll
      for (int p = 0; p < 8; ++p) {
        // B 32x16: lanes 0-15 -> K {0..7,16..23}; lanes 16-31 -> +8
        int kk = ((p < 4) ? (2 * p) : (2 * p + 8)) + grp * 8;
        fb[j].u[p] = *(const unsigned int*)&Bs[bcol * BS_STR + kk];
      }
    }
    #pragma unroll
    for (int i = 0; i < 2; ++i)
      #pragma unroll
      for (int j = 0; j < 2; ++j)
        acc[i][j] = __builtin_amdgcn_wmma_f32_16x16x32_f16(
            false, fa[i].v, false, fb[j].v, (short)0, acc[i][j], false, false);
    __syncthreads();
  }

  // Epilogue. C/D layout: VGPR r -> M = r + 8*grp, N = lane&15.
  // Fast path for interior blocks: no guards, one base pointer, immediate
  // column offsets for the second j-tile.
  const int rbase = row0 + wm * 32 + 8 * grp;
  const int gcol  = col0 + wn * 32 + l16;
  float* cp = C + (size_t)rbase * Nt + gcol;
  if (row0 + BM <= M) {
    #pragma unroll
    for (int i = 0; i < 2; ++i)
      #pragma unroll
      for (int r = 0; r < 8; ++r) {
        float* q = cp + (size_t)(i * 16 + r) * Nt;
        q[0]  = acc[i][0][r];
        q[16] = acc[i][1][r];
      }
  } else {
    #pragma unroll
    for (int i = 0; i < 2; ++i)
      #pragma unroll
      for (int r = 0; r < 8; ++r) {
        if (rbase + i * 16 + r < M) {
          float* q = cp + (size_t)(i * 16 + r) * Nt;
          q[0]  = acc[i][0][r];
          q[16] = acc[i][1][r];
        }
      }
  }
}

// ---------------------------------------------------------------------------
// alpha_s / alpha_d: one wave32 per (node, head); dual dot + shuffle reduce.
// heads is a power of two -> hshift.
// ---------------------------------------------------------------------------
__global__ __launch_bounds__(256) void gat_alpha(
    const float* __restrict__ xw, const float* __restrict__ a_src,
    const float* __restrict__ a_dst, float* __restrict__ as_out,
    float* __restrict__ ad_out, int n_nodes, int hshift, int ch)
{
  int wave = blockIdx.x * (blockDim.x >> 5) + (threadIdx.x >> 5);
  int lane = threadIdx.x & 31;
  int n = wave >> hshift, h = wave & ((1 << hshift) - 1);
  if (n >= n_nodes) return;
  const float* xr  = xw + (((size_t)n << hshift) + h) * ch;
  const float* asr = a_src + h * ch;
  const float* adr = a_dst + h * ch;
  float s = 0.f, d = 0.f;
  for (int c = lane; c < ch; c += 32) {
    float x = xr[c];
    s += x * asr[c];
    d += x * adr[c];
  }
  for (int off = 16; off > 0; off >>= 1) {
    s += __shfl_down(s, off, 32);
    d += __shfl_down(d, off, 32);
  }
  if (lane == 0) {
    as_out[(n << hshift) + h] = s;
    ad_out[(n << hshift) + h] = d;
  }
}

// ---------------------------------------------------------------------------
// Edge helpers: self-loops synthesized for e >= E; monotone float->uint keys
// so segment-max can use hardware atomicMax(u32).
// ---------------------------------------------------------------------------
__device__ __forceinline__ void edge_sd(const int* ei, int e, int& s, int& d) {
  if (e < N_EDGES) { s = ei[e]; d = ei[N_EDGES + e]; }
  else { s = e - N_EDGES; d = s; }
}
__device__ __forceinline__ unsigned int fkey(float f) {
  unsigned int u = __float_as_uint(f);
  return (u & 0x80000000u) ? ~u : (u | 0x80000000u);
}
__device__ __forceinline__ float funkey(unsigned int k) {
  return __uint_as_float((k & 0x80000000u) ? (k & 0x7fffffffu) : ~k);
}

__global__ void gat_edge_logit_max(const int* __restrict__ ei,
    const float* __restrict__ as, const float* __restrict__ ad,
    float* __restrict__ elog, unsigned int* __restrict__ mkey, int hshift)
{
  int idx = blockIdx.x * blockDim.x + threadIdx.x;
  if (idx >= (ETOT << hshift)) return;
  int e = idx >> hshift, h = idx & ((1 << hshift) - 1);
  int s, d; edge_sd(ei, e, s, d);
  float l = as[(s << hshift) + h] + ad[(d << hshift) + h];
  l = (l > 0.f) ? l : NEG_SLOPE * l;                 // leaky_relu
  elog[idx] = l;
  atomicMax(&mkey[(d << hshift) + h], fkey(l));
}

__global__ void gat_edge_exp(const int* __restrict__ ei,
    float* __restrict__ elog, const unsigned int* __restrict__ mkey,
    float* __restrict__ denom, int hshift)
{
  int idx = blockIdx.x * blockDim.x + threadIdx.x;
  if (idx >= (ETOT << hshift)) return;
  int e = idx >> hshift, h = idx & ((1 << hshift) - 1);
  int s, d; edge_sd(ei, e, s, d);
  float ex = __expf(elog[idx] - funkey(mkey[(d << hshift) + h]));
  elog[idx] = ex;
  atomicAdd(&denom[(d << hshift) + h], ex);
}

__global__ __launch_bounds__(256) void gat_edge_aggregate(
    const int* __restrict__ ei, const float* __restrict__ xw,
    const float* __restrict__ elog, const float* __restrict__ denom,
    float* __restrict__ out, int hshift, int cshift)
{
  int e = blockIdx.x;
  int s, d; edge_sd(ei, e, s, d);
  int hc = 1 << (hshift + cshift);
  for (int c = threadIdx.x; c < hc; c += blockDim.x) {
    int h = c >> cshift;
    float coef = elog[(e << hshift) + h] / denom[(d << hshift) + h];
    atomicAdd(&out[((size_t)d << (hshift + cshift)) + c],
              coef * xw[((size_t)s << (hshift + cshift)) + c]);
  }
}

__global__ void gat_bias_elu(float* __restrict__ x, const float* __restrict__ b,
                             int hcmask, long long total)
{
  long long idx = (long long)blockIdx.x * blockDim.x + threadIdx.x;
  if (idx >= total) return;
  float v = x[idx] + b[(int)idx & hcmask];
  x[idx] = (v > 0.f) ? v : (__expf(v) - 1.f);       // elu
}

__global__ __launch_bounds__(256) void gat_mean(const float* __restrict__ out2,
    const float* __restrict__ b2, float* __restrict__ g, int n_nodes, int ch)
{
  __shared__ float red[256];
  int c = blockIdx.x;
  float s = 0.f;
  for (int n = threadIdx.x; n < n_nodes; n += blockDim.x)
    s += out2[(size_t)n * ch + c];
  red[threadIdx.x] = s;
  __syncthreads();
  for (int off = 128; off > 0; off >>= 1) {
    if (threadIdx.x < off) red[threadIdx.x] += red[threadIdx.x + off];
    __syncthreads();
  }
  if (threadIdx.x == 0) g[c] = red[0] / (float)n_nodes + b2[c];
}

// ---------------------------------------------------------------------------
extern "C" void kernel_launch(void* const* d_in, const int* in_sizes, int n_in,
                              void* d_out, int out_size, void* d_ws, size_t ws_size,
                              hipStream_t stream) {
  (void)in_sizes; (void)n_in; (void)out_size; (void)ws_size;
  const int*   ei   = (const int*)d_in[0];
  const float* x0   = (const float*)d_in[1];
  const float* W1   = (const float*)d_in[2];
  const float* a_s1 = (const float*)d_in[3];
  const float* a_d1 = (const float*)d_in[4];
  const float* b1   = (const float*)d_in[5];
  const float* W2   = (const float*)d_in[6];
  const float* a_s2 = (const float*)d_in[7];
  const float* a_d2 = (const float*)d_in[8];
  const float* b2   = (const float*)d_in[9];
  float* out = (float*)d_out;

  char* p = (char*)d_ws;
  auto carve = [&](size_t bytes) {
    char* q = p; p += (bytes + 255) & ~(size_t)255; return q;
  };
  float* xw1  = (float*)carve((size_t)N_NODES * HEADS * HID * sizeof(float)); // later: xw2 | out2
  float* out1 = (float*)carve((size_t)N_NODES * HEADS * HID * sizeof(float)); // -> x1 after ELU
  float* elog = (float*)carve((size_t)ETOT * HEADS * sizeof(float));
  float* as_b = (float*)carve((size_t)N_NODES * HEADS * sizeof(float));
  float* ad_b = (float*)carve((size_t)N_NODES * HEADS * sizeof(float));
  unsigned int* mkey = (unsigned int*)carve((size_t)N_NODES * HEADS * sizeof(unsigned int));
  float* denom = (float*)carve((size_t)N_NODES * HEADS * sizeof(float));
  float* xw2  = xw1;                                   // N x 256
  float* out2 = xw1 + (size_t)N_NODES * OUT_DIM;       // N x 256 (disjoint halves)

  // -------------------- Layer 1: GATConv(64 -> 128, heads=4) --------------
  {
    dim3 grid((HEADS * HID) / BN, (N_NODES + BM - 1) / BM);
    gat_wmma_gemm<<<grid, 256, 0, stream>>>(x0, W1, xw1, N_NODES, F_IN, HEADS * HID);
  }
  gat_alpha<<<(N_NODES * HEADS + 7) / 8, 256, 0, stream>>>(
      xw1, a_s1, a_d1, as_b, ad_b, N_NODES, /*hshift=*/2, HID);

  hipMemsetAsync(mkey, 0, (size_t)N_NODES * HEADS * 4, stream);
  hipMemsetAsync(denom, 0, (size_t)N_NODES * HEADS * 4, stream);
  hipMemsetAsync(out1, 0, (size_t)N_NODES * HEADS * HID * 4, stream);

  {
    int total = ETOT * HEADS;
    gat_edge_logit_max<<<(total + 255) / 256, 256, 0, stream>>>(ei, as_b, ad_b, elog, mkey, 2);
    gat_edge_exp<<<(total + 255) / 256, 256, 0, stream>>>(ei, elog, mkey, denom, 2);
    gat_edge_aggregate<<<ETOT, 256, 0, stream>>>(ei, xw1, elog, denom, out1,
                                                 /*hshift=*/2, /*cshift=*/7);
  }
  {
    long long total = (long long)N_NODES * HEADS * HID;
    gat_bias_elu<<<(unsigned)((total + 255) / 256), 256, 0, stream>>>(
        out1, b1, HEADS * HID - 1, total);
  }

  // -------------------- Layer 2: GATConv(512 -> 256, heads=1) -------------
  {
    dim3 grid(OUT_DIM / BN, (N_NODES + BM - 1) / BM);
    gat_wmma_gemm<<<grid, 256, 0, stream>>>(out1, W2, xw2, N_NODES, HEADS * HID, OUT_DIM);
  }
  gat_alpha<<<(N_NODES + 7) / 8, 256, 0, stream>>>(
      xw2, a_s2, a_d2, as_b, ad_b, N_NODES, /*hshift=*/0, OUT_DIM);

  hipMemsetAsync(mkey, 0, (size_t)N_NODES * 4, stream);
  hipMemsetAsync(denom, 0, (size_t)N_NODES * 4, stream);
  hipMemsetAsync(out2, 0, (size_t)N_NODES * OUT_DIM * 4, stream);

  {
    gat_edge_logit_max<<<(ETOT + 255) / 256, 256, 0, stream>>>(ei, as_b, ad_b, elog, mkey, 0);
    gat_edge_exp<<<(ETOT + 255) / 256, 256, 0, stream>>>(ei, elog, mkey, denom, 0);
    gat_edge_aggregate<<<ETOT, 256, 0, stream>>>(ei, xw2, elog, denom, out2,
                                                 /*hshift=*/0, /*cshift=*/8);
  }

  gat_mean<<<OUT_DIM, 256, 0, stream>>>(out2, b2, out, N_NODES, OUT_DIM);
}
